// CrossAttention_52484500357469
// MI455X (gfx1250) — matrix operand order repaired and checked by reading, not compiled, not run on hardware
//
#include <hip/hip_runtime.h>
#include <hip/hip_bf16.h>

typedef __attribute__((ext_vector_type(16))) __bf16 v16bf;
typedef __attribute__((ext_vector_type(8)))  __bf16 v8bf;
typedef __attribute__((ext_vector_type(8)))  float  v8f;

#define WMMA_BF16(a, b, c) \
    __builtin_amdgcn_wmma_f32_16x16x32_bf16(false, (a), false, (b), (short)0, (c), false, false)

// Async DMA: copy 16 bytes/lane from global to LDS (ASYNCcnt-tracked).
__device__ __forceinline__ void async_b128(void* lds_ptr, const void* gptr)
{
    unsigned loff = (unsigned)(size_t)lds_ptr;   // low 32 bits of flat addr = LDS offset
    asm volatile("global_load_async_to_lds_b128 %0, %1, off"
                 :: "v"(loff), "v"(gptr) : "memory");
}
__device__ __forceinline__ void wait_async0()
{
    asm volatile("s_wait_asynccnt 0x0" ::: "memory");
}

// One xor-butterfly max step within each 16-lane row: pure VALU via
// v_permlane16_b32 (static lane-select pattern), no LDS round trip.
__device__ __forceinline__ float rowmax_step(float x, unsigned s1, unsigned s2)
{
    union { float f; unsigned u; } a, b;
    a.f = x;
    b.u = __builtin_amdgcn_permlane16(a.u, a.u, s1, s2, false, false);
    return fmaxf(x, b.f);
}

// ---------------------------------------------------------------------------
// One-time weight transpose + f32->bf16 convert (tiny, L2-resident after).
// W[K][Nc] f32  ->  WT[Nc][K] bf16, scaled.
// ---------------------------------------------------------------------------
__global__ void transpose_w(const float* __restrict__ W, __bf16* __restrict__ WT,
                            int K, int Nc, float scale)
{
    int idx = blockIdx.x * blockDim.x + threadIdx.x;
    if (idx >= K * Nc) return;
    int kk = idx / Nc, n = idx % Nc;
    WT[(size_t)n * K + kk] = (__bf16)(W[idx] * scale);
}

// ---------------------------------------------------------------------------
// WMMA GEMM, no LDS: C[M,N] = A[M,K] * WT^T (+bias).  WT is [N][K] bf16 so
// B-fragments (kk-contiguous per output column) are two 128-bit global loads.
// Each wave: one 16-row M tile x two 16-col N tiles (A-fragment reuse).
// OMODE: 0 = bf16 row-major out, 1 = bf16 transposed out, 2 = f32 + bias out.
// ---------------------------------------------------------------------------
template <bool A_F32, int OMODE>
__global__ void __launch_bounds__(128)
gemm_wmma(const void* __restrict__ A_, const __bf16* __restrict__ WT,
          void* __restrict__ out_, const float* __restrict__ bias,
          int Mrows, int Kdim, int Ncols)
{
    const int wave = threadIdx.x >> 5;
    const int lane = threadIdx.x & 31;
    const int half = lane >> 4;
    const int l16  = lane & 15;

    const int n_base = blockIdx.x * 32;
    const int m_base = (blockIdx.y * 4 + wave) * 16;

    const float*  Af = (const float*)A_;
    const __bf16* Ab = (const __bf16*)A_;

    const size_t  arow = (size_t)(m_base + l16) * Kdim;
    const __bf16* w0 = WT + (size_t)(n_base + l16) * Kdim + half * 16;
    const __bf16* w1 = WT + (size_t)(n_base + 16 + l16) * Kdim + half * 16;

    v8f acc0 = {}, acc1 = {};

    for (int k0 = 0; k0 < Kdim; k0 += 32) {
        v16bf a, b0, b1;
        if (A_F32) {
            v8f lo = *(const v8f*)(Af + arow + k0 + half * 16);
            v8f hi = *(const v8f*)(Af + arow + k0 + half * 16 + 8);
            #pragma unroll
            for (int e = 0; e < 8; ++e) { a[e] = (__bf16)lo[e]; a[8 + e] = (__bf16)hi[e]; }
        } else {
            v8bf lo = *(const v8bf*)(Ab + arow + k0 + half * 16);
            v8bf hi = *(const v8bf*)(Ab + arow + k0 + half * 16 + 8);
            #pragma unroll
            for (int e = 0; e < 8; ++e) { a[e] = lo[e]; a[8 + e] = hi[e]; }
        }
        {
            v8bf x0 = *(const v8bf*)(w0 + k0);
            v8bf x1 = *(const v8bf*)(w0 + k0 + 8);
            v8bf y0 = *(const v8bf*)(w1 + k0);
            v8bf y1 = *(const v8bf*)(w1 + k0 + 8);
            #pragma unroll
            for (int e = 0; e < 8; ++e) {
                b0[e] = x0[e]; b0[8 + e] = x1[e];
                b1[e] = y0[e]; b1[8 + e] = y1[e];
            }
        }
        acc0 = WMMA_BF16(a, b0, acc0);
        acc1 = WMMA_BF16(a, b1, acc1);
    }

    const int nc0 = n_base + l16;
    const int nc1 = n_base + 16 + l16;
    if (OMODE == 2) {
        float* Of = (float*)out_;
        float bb0 = bias[nc0], bb1 = bias[nc1];
        #pragma unroll
        for (int r = 0; r < 8; ++r) {
            int row = m_base + r + half * 8;
            Of[(size_t)row * Ncols + nc0] = acc0[r] + bb0;
            Of[(size_t)row * Ncols + nc1] = acc1[r] + bb1;
        }
    } else if (OMODE == 1) {
        // Transposed bf16 out: OT[n][m]; the 8 rows per lane are consecutive m.
        __bf16* Ob = (__bf16*)out_;
        v8bf o0, o1;
        #pragma unroll
        for (int r = 0; r < 8; ++r) { o0[r] = (__bf16)acc0[r]; o1[r] = (__bf16)acc1[r]; }
        *(v8bf*)(Ob + (size_t)nc0 * Mrows + m_base + half * 8) = o0;
        *(v8bf*)(Ob + (size_t)nc1 * Mrows + m_base + half * 8) = o1;
    } else {
        __bf16* Ob = (__bf16*)out_;
        #pragma unroll
        for (int r = 0; r < 8; ++r) {
            int row = m_base + r + half * 8;
            Ob[(size_t)row * Ncols + nc0] = (__bf16)acc0[r];
            Ob[(size_t)row * Ncols + nc1] = (__bf16)acc1[r];
        }
    }
}

// ---------------------------------------------------------------------------
// Flash attention. grid (N/64, H, B), 4 waves/block. Softmax scale is baked
// into q. K/V 32-key tiles are DMA'd into double-buffered LDS with
// global_load_async_to_lds_b128 (next tile prefetched while current computes);
// fragments come from LDS via 128-bit ds loads. Row sums of P via WMMA
// against an all-ones B fragment; row max via permlane16 VALU butterflies.
// ---------------------------------------------------------------------------
__global__ void __launch_bounds__(128)
attn_wmma(const __bf16* __restrict__ q, const __bf16* __restrict__ k,
          const __bf16* __restrict__ vT, __bf16* __restrict__ o,
          int N, int M, int BM)
{
    // Padded rows: 16B-aligned row bases, banks spread across the 64 banks.
    __shared__ __align__(16) __bf16 Ks[2][32][72];   // [buf][key][d]
    __shared__ __align__(16) __bf16 Vs[2][64][40];   // [buf][d][key]
    __shared__ __align__(16) __bf16 Ps[4][16][32];   // per-wave P tile

    const int tid  = threadIdx.x;
    const int wave = tid >> 5;
    const int lane = tid & 31;
    const int half = lane >> 4;
    const int l16  = lane & 15;

    const int h = blockIdx.y, b = blockIdx.z;
    const int qrow0 = b * N + blockIdx.x * 64 + wave * 16;
    const int krow0 = b * M;
    const int D = 512, hd = h * 64;

    // Issue the async DMA for one 32-key K tile + V tile into buffer `buf`.
    auto stage = [&](int buf, int j) {
        #pragma unroll
        for (int it = 0; it < 2; ++it) {
            int chunk = tid + it * 128;              // 256 x 16B = 4KB per tile
            int key = chunk >> 3, dby = (chunk & 7) * 16;    // K rows: 128B
            async_b128((char*)&Ks[buf][key][0] + dby,
                       (const char*)k + ((size_t)(krow0 + j + key) * D + hd) * 2 + dby);
            int d = chunk >> 2, kby = (chunk & 3) * 16;      // V rows: 64B
            async_b128((char*)&Vs[buf][d][0] + kby,
                       (const char*)vT + ((size_t)(hd + d) * BM + krow0 + j) * 2 + kby);
        }
    };

    // Q A-fragments (scale already folded into Wq^T).
    v16bf aq[2];
    {
        const __bf16* qp = q + (size_t)(qrow0 + l16) * D + hd;
        #pragma unroll
        for (int c2 = 0; c2 < 2; ++c2) {
            v8bf lo = *(const v8bf*)(qp + c2 * 32 + half * 16);
            v8bf hi = *(const v8bf*)(qp + c2 * 32 + half * 16 + 8);
            #pragma unroll
            for (int e = 0; e < 8; ++e) { aq[c2][e] = lo[e]; aq[c2][8 + e] = hi[e]; }
        }
    }

    v16bf ones;
    #pragma unroll
    for (int e = 0; e < 16; ++e) ones[e] = (__bf16)1.0f;

    v8f vz = {};
    float m_prev[8], l_sum[8];
    v8f acc[4];
    #pragma unroll
    for (int r = 0; r < 8; ++r) { m_prev[r] = -1e30f; l_sum[r] = 0.0f; }
    #pragma unroll
    for (int c = 0; c < 4; ++c) acc[c] = vz;

    stage(0, 0);   // prime the pipeline

    for (int j0 = 0; j0 < M; j0 += 32) {
        const int buf = (j0 >> 5) & 1;

        wait_async0();       // our DMA into `buf` has landed
        __syncthreads();     // everyone's DMA landed; prior reads of buf^1 done
        if (j0 + 32 < M) stage(buf ^ 1, j0 + 32);   // overlap next tile's DMA

        // --- S = (scaled Q) K^T, fragments from LDS.
        v16bf bk[2][2];
        #pragma unroll
        for (int g = 0; g < 2; ++g) {
            #pragma unroll
            for (int c2 = 0; c2 < 2; ++c2) {
                const __bf16* p = &Ks[buf][g * 16 + l16][c2 * 32 + half * 16];
                v8bf x0 = *(const v8bf*)p;
                v8bf x1 = *(const v8bf*)(p + 8);
                #pragma unroll
                for (int e = 0; e < 8; ++e) { bk[g][c2][e] = x0[e]; bk[g][c2][8 + e] = x1[e]; }
            }
        }
        v8f s0 = vz, s1 = vz;
        s0 = WMMA_BF16(aq[0], bk[0][0], s0);
        s0 = WMMA_BF16(aq[1], bk[0][1], s0);
        s1 = WMMA_BF16(aq[0], bk[1][0], s1);
        s1 = WMMA_BF16(aq[1], bk[1][1], s1);

        // --- row max: 4 permlane16 butterfly steps, 8 independent rows each.
        float rm[8];
        #pragma unroll
        for (int r = 0; r < 8; ++r) rm[r] = fmaxf(s0[r], s1[r]);
        #pragma unroll
        for (int r = 0; r < 8; ++r) rm[r] = rowmax_step(rm[r], 0x67452301u, 0xEFCDAB89u); // ^1
        #pragma unroll
        for (int r = 0; r < 8; ++r) rm[r] = rowmax_step(rm[r], 0x54761032u, 0xDCFE98BAu); // ^2
        #pragma unroll
        for (int r = 0; r < 8; ++r) rm[r] = rowmax_step(rm[r], 0x32107654u, 0xBA98FEDCu); // ^4
        #pragma unroll
        for (int r = 0; r < 8; ++r) rm[r] = rowmax_step(rm[r], 0xFEDCBA98u, 0x76543210u); // ^8

        // --- online softmax update; write P tile (bf16) to per-wave LDS.
        float alpha[8];
        #pragma unroll
        for (int r = 0; r < 8; ++r) {
            float m_new = fmaxf(m_prev[r], rm[r]);
            alpha[r] = __expf(m_prev[r] - m_new);
            float p0 = __expf(s0[r] - m_new);
            float p1 = __expf(s1[r] - m_new);
            m_prev[r] = m_new;
            int row = r + half * 8;
            Ps[wave][row][l16]      = (__bf16)p0;
            Ps[wave][row][16 + l16] = (__bf16)p1;
            #pragma unroll
            for (int c = 0; c < 4; ++c) acc[c][r] *= alpha[r];
        }

        // --- P A-fragment (C-layout -> A-layout via LDS; in-order per wave).
        v16bf ap;
        {
            const __bf16* pr = &Ps[wave][l16][0];
            v8bf lo = *(const v8bf*)(pr + half * 16);
            v8bf hi = *(const v8bf*)(pr + half * 16 + 8);
            #pragma unroll
            for (int e = 0; e < 8; ++e) { ap[e] = lo[e]; ap[8 + e] = hi[e]; }
        }

        // --- row sums of P via WMMA against all-ones B (replaces shuffles).
        v8f rs = WMMA_BF16(ap, ones, vz);
        #pragma unroll
        for (int r = 0; r < 8; ++r) l_sum[r] = l_sum[r] * alpha[r] + rs[r];

        // --- O += P * V, fragments from LDS.
        v16bf bv[4];
        #pragma unroll
        for (int c = 0; c < 4; ++c) {
            const __bf16* p = &Vs[buf][c * 16 + l16][half * 16];
            v8bf x0 = *(const v8bf*)p;
            v8bf x1 = *(const v8bf*)(p + 8);
            #pragma unroll
            for (int e = 0; e < 8; ++e) { bv[c][e] = x0[e]; bv[c][8 + e] = x1[e]; }
        }
        #pragma unroll
        for (int c = 0; c < 4; ++c) acc[c] = WMMA_BF16(ap, bv[c], acc[c]);
    }

    // Normalize and store bf16 attention output [B*N][512].
    #pragma unroll
    for (int r = 0; r < 8; ++r) {
        int   row = qrow0 + r + half * 8;
        float inv = 1.0f / l_sum[r];
        #pragma unroll
        for (int c = 0; c < 4; ++c)
            o[(size_t)row * D + hd + c * 16 + l16] = (__bf16)(acc[c][r] * inv);
    }
}

// ---------------------------------------------------------------------------
extern "C" void kernel_launch(void* const* d_in, const int* in_sizes, int n_in,
                              void* d_out, int out_size, void* d_ws, size_t ws_size,
                              hipStream_t stream)
{
    const float* x   = (const float*)d_in[0];
    const float* ctx = (const float*)d_in[1];
    const float* Wq  = (const float*)d_in[2];
    const float* Wk  = (const float*)d_in[3];
    const float* Wv  = (const float*)d_in[4];
    const float* Wo  = (const float*)d_in[5];
    const float* bo  = (const float*)d_in[6];
    float* out = (float*)d_out;

    const int B = 2, N = 4096, Mc = 1024, QDIM = 320, CDIM = 768, INNER = 512, H = 8;
    const int BN = B * N, BMc = B * Mc;

    // Workspace (bf16 elements)
    __bf16* qb  = (__bf16*)d_ws;                     // [BN][512]
    __bf16* kb  = qb  + (size_t)BN * INNER;          // [BMc][512]
    __bf16* vtb = kb  + (size_t)BMc * INNER;         // [512][BMc]  (transposed V)
    __bf16* ob  = vtb + (size_t)BMc * INNER;         // [BN][512]
    __bf16* WqT = ob  + (size_t)BN * INNER;          // [512][320]
    __bf16* WkT = WqT + (size_t)INNER * QDIM;        // [512][768]
    __bf16* WvT = WkT + (size_t)INNER * CDIM;        // [512][768]
    __bf16* WoT = WvT + (size_t)INNER * CDIM;        // [320][512]

    // Weight transposes (scale 1/sqrt(64)=0.125 baked into Wq^T; exact in bf16)
    transpose_w<<<(QDIM * INNER + 255) / 256, 256, 0, stream>>>(Wq, WqT, QDIM, INNER, 0.125f);
    transpose_w<<<(CDIM * INNER + 255) / 256, 256, 0, stream>>>(Wk, WkT, CDIM, INNER, 1.0f);
    transpose_w<<<(CDIM * INNER + 255) / 256, 256, 0, stream>>>(Wv, WvT, CDIM, INNER, 1.0f);
    transpose_w<<<(INNER * QDIM + 255) / 256, 256, 0, stream>>>(Wo, WoT, INNER, QDIM, 1.0f);

    dim3 blk(128);
    // Projections (f32 A -> bf16 out)
    gemm_wmma<true, 0><<<dim3(INNER / 32, BN / 64), blk, 0, stream>>>(
        x, WqT, qb, nullptr, BN, QDIM, INNER);
    gemm_wmma<true, 0><<<dim3(INNER / 32, BMc / 64), blk, 0, stream>>>(
        ctx, WkT, kb, nullptr, BMc, CDIM, INNER);
    gemm_wmma<true, 1><<<dim3(INNER / 32, BMc / 64), blk, 0, stream>>>(
        ctx, WvT, vtb, nullptr, BMc, CDIM, INNER);

    // Attention (async-DMA double-buffered K/V tiles)
    attn_wmma<<<dim3(N / 64, H, B), blk, 0, stream>>>(qb, kb, vtb, ob, N, Mc, BMc);

    // Output projection + bias (bf16 A -> f32 out)
    gemm_wmma<false, 2><<<dim3(QDIM / 32, BN / 64), blk, 0, stream>>>(
        ob, WoT, out, bo, BN, INNER, QDIM);
}
